// AttentionRAR_75677323755909
// MI455X (gfx1250) — matrix operand order repaired and checked by reading, not compiled, not run on hardware
//
#include <hip/hip_runtime.h>

// ---------------------------------------------------------------------------
// MHA forward for B=4, N=2048, C=1024, H=16, D=64 on gfx1250 (MI455X).
// All GEMMs via v_wmma_f32_16x16x32_bf16 (wave32). Flash-attention style
// streaming softmax so the [B,H,N,N] score tensor never hits memory.
// ---------------------------------------------------------------------------

typedef __attribute__((ext_vector_type(16))) __bf16 v16bf;
typedef __attribute__((ext_vector_type(8)))  __bf16 v8bf;
typedef __attribute__((ext_vector_type(8)))  float  v8f;

#define WMMA_BF16(A, B, C) \
  __builtin_amdgcn_wmma_f32_16x16x32_bf16(false, (A), false, (B), (short)0, (C), false, false)

// A fragment 16x32 bf16 (row-major source): lane group g holds K
// [16g..16g+15] contiguous -> two contiguous 16B loads.
__device__ __forceinline__ v16bf load_a16(const __bf16* rowp, int k0, int g) {
  v8bf lo = *(const v8bf*)(rowp + k0 + g * 16);
  v8bf hi = *(const v8bf*)(rowp + k0 + g * 16 + 8);
  v16bf a;
#pragma unroll
  for (int i = 0; i < 8; ++i) { a[i] = lo[i]; a[i + 8] = hi[i]; }
  return a;
}

// B fragment 32x16 bf16 (column-major source): lane holds col = lane%16,
// K = 16g..16g+15 contiguous -> one 32B load.
__device__ __forceinline__ v16bf load_b16(const __bf16* colp, int k0, int g) {
  return *(const v16bf*)(colp + k0 + g * 16);
}

// ------------------------- precision conversion ----------------------------
__global__ void cvt_f32_bf16(const float* __restrict__ src,
                             __bf16* __restrict__ dst, int n) {
  int i = blockIdx.x * blockDim.x + threadIdx.x;
  int stride = gridDim.x * blockDim.x;
  for (; i < n; i += stride) dst[i] = (__bf16)src[i];
}

// src [rows][cols] f32 -> dst [cols][rows] bf16 (tiled via LDS)
__global__ __launch_bounds__(256) void transpose_f32_to_bf16(
    const float* __restrict__ src, __bf16* __restrict__ dst, int rows, int cols) {
  __shared__ float tile[32][33];
  const int c0 = blockIdx.x * 32, r0 = blockIdx.y * 32;
#pragma unroll
  for (int j = 0; j < 32; j += 8)
    tile[threadIdx.y + j][threadIdx.x] =
        src[(size_t)(r0 + threadIdx.y + j) * cols + c0 + threadIdx.x];
  __syncthreads();
#pragma unroll
  for (int j = 0; j < 32; j += 8)
    dst[(size_t)(c0 + threadIdx.y + j) * rows + r0 + threadIdx.x] =
        (__bf16)tile[threadIdx.x][threadIdx.y + j];
}

// ------------------------------ QKV GEMM -----------------------------------
// xb [8192,1024] bf16  x  wT [3072(col),1024(k)] bf16 -> q (pre-scaled 1/8),
// k in [B,H,N,D], v transposed into [B,H,D,N].  Wave tile: 64 rows x 32 cols.
__global__ __launch_bounds__(128) void qkv_gemm(
    const __bf16* __restrict__ xb, const __bf16* __restrict__ wT,
    __bf16* __restrict__ qd, __bf16* __restrict__ kd, __bf16* __restrict__ vtd) {
  const int lane = threadIdx.x & 31, wv = threadIdx.x >> 5;
  const int g = lane >> 4, ln = lane & 15;
  const int rowbase = blockIdx.x * 64;
  const int colbase = blockIdx.y * 128 + wv * 32;

  v8f acc[4][2];
#pragma unroll
  for (int ti = 0; ti < 4; ++ti)
#pragma unroll
    for (int tj = 0; tj < 2; ++tj) acc[ti][tj] = (v8f)0.0f;

  const __bf16* ar[4];
  const __bf16* bc[2];
#pragma unroll
  for (int ti = 0; ti < 4; ++ti)
    ar[ti] = xb + (size_t)(rowbase + ti * 16 + ln) * 1024;
#pragma unroll
  for (int tj = 0; tj < 2; ++tj)
    bc[tj] = wT + (size_t)(colbase + tj * 16 + ln) * 1024;

  for (int k0 = 0; k0 < 1024; k0 += 32) {
    v16bf a[4], b[2];
#pragma unroll
    for (int ti = 0; ti < 4; ++ti) a[ti] = load_a16(ar[ti], k0, g);
#pragma unroll
    for (int tj = 0; tj < 2; ++tj) b[tj] = load_b16(bc[tj], k0, g);
#pragma unroll
    for (int ti = 0; ti < 4; ++ti)
#pragma unroll
      for (int tj = 0; tj < 2; ++tj)
        acc[ti][tj] = WMMA_BF16(a[ti], b[tj], acc[ti][tj]);
  }

#pragma unroll
  for (int ti = 0; ti < 4; ++ti)
#pragma unroll
    for (int tj = 0; tj < 2; ++tj)
#pragma unroll
      for (int r = 0; r < 8; ++r) {
        const int grow = rowbase + ti * 16 + r + (g << 3);
        const int gcol = colbase + tj * 16 + ln;
        const int which = gcol >> 10, cc = gcol & 1023;
        const int h = cc >> 6, d = cc & 63;
        const int b_ = grow >> 11, n = grow & 2047;
        const size_t bh = (size_t)(b_ * 16 + h);
        const float v = acc[ti][tj][r];
        if (which == 0)       qd [(bh * 2048 + n) * 64 + d] = (__bf16)(v * 0.125f);
        else if (which == 1)  kd [(bh * 2048 + n) * 64 + d] = (__bf16)v;
        else                  vtd[(bh * 64 + d) * 2048 + n] = (__bf16)v;
      }
}

// --------------------------- flash attention -------------------------------
// One wave = one 16-row query tile of one (b,h). Streams 32-key tiles up to
// the causal diagonal; additive mask from input applied on the last tile.
__global__ __launch_bounds__(128) void attn_kernel(
    const __bf16* __restrict__ q, const __bf16* __restrict__ k,
    const __bf16* __restrict__ vt, const float* __restrict__ mask,
    __bf16* __restrict__ attb) {
  __shared__ __bf16 plds[4][16 * 40];  // per-wave 16x32 P tile, 40-elt (80B) row stride
  const int lane = threadIdx.x & 31, wv = threadIdx.x >> 5;
  const int g = lane >> 4, ln = lane & 15;
  const int bh = blockIdx.x >> 5;                 // 0..63
  const int qt = ((blockIdx.x & 31) << 2) | wv;   // 0..127
  const int qbase = qt * 16;

  const __bf16* Q  = q  + (size_t)bh * 2048 * 64;
  const __bf16* Km = k  + (size_t)bh * 2048 * 64;
  const __bf16* Vt = vt + (size_t)bh * 64 * 2048;

  const __bf16* qrow = Q + (size_t)(qbase + ln) * 64;
  const v16bf qa0 = load_a16(qrow, 0, g);   // d = 0..31 (Q pre-scaled by 1/8)
  const v16bf qa1 = load_a16(qrow, 32, g);  // d = 32..63

  v8f O[4];
#pragma unroll
  for (int t = 0; t < 4; ++t) O[t] = (v8f)0.0f;
  float rmax[8], rsum[8];
#pragma unroll
  for (int r = 0; r < 8; ++r) { rmax[r] = -3.0e38f; rsum[r] = 0.0f; }

  __bf16* P = &plds[wv][0];
  const int nkt = ((qbase + 15) >> 5) + 1;

  for (int kt = 0; kt < nkt; ++kt) {
    const int kb = kt << 5;
    // ---- S = Q * K^T  (two 16x16 score tiles, K rows are B-frag columns)
    const __bf16* kc0 = Km + (size_t)(kb + ln) * 64;
    const __bf16* kc1 = kc0 + 16 * 64;
    v8f S0 = (v8f)0.0f, S1 = (v8f)0.0f;
    S0 = WMMA_BF16(qa0, load_b16(kc0, 0, g),  S0);
    S0 = WMMA_BF16(qa1, load_b16(kc0, 32, g), S0);
    S1 = WMMA_BF16(qa0, load_b16(kc1, 0, g),  S1);
    S1 = WMMA_BF16(qa1, load_b16(kc1, 32, g), S1);

    const bool lastTile = (kt == nkt - 1);
#pragma unroll
    for (int r = 0; r < 8; ++r) {
      float s0 = S0[r], s1 = S1[r];
      if (lastTile) {  // additive causal mask only straddles the final tile
        const int n = qbase + r + (g << 3);
        s0 += mask[(size_t)n * 2048 + kb + ln];
        s1 += mask[(size_t)n * 2048 + kb + 16 + ln];
      }
      // row max across the 16 lanes of this lane-group (columns of the row)
      float mt = fmaxf(s0, s1);
      mt = fmaxf(mt, __shfl_xor(mt, 1, 32));
      mt = fmaxf(mt, __shfl_xor(mt, 2, 32));
      mt = fmaxf(mt, __shfl_xor(mt, 4, 32));
      mt = fmaxf(mt, __shfl_xor(mt, 8, 32));
      const float mnew  = fmaxf(rmax[r], mt);
      const float alpha = __expf(rmax[r] - mnew);
      rmax[r] = mnew;
      const float p0 = __expf(s0 - mnew);
      const float p1 = __expf(s1 - mnew);
      float ps = p0 + p1;
      ps += __shfl_xor(ps, 1, 32);
      ps += __shfl_xor(ps, 2, 32);
      ps += __shfl_xor(ps, 4, 32);
      ps += __shfl_xor(ps, 8, 32);
      rsum[r] = rsum[r] * alpha + ps;
      O[0][r] *= alpha; O[1][r] *= alpha; O[2][r] *= alpha; O[3][r] *= alpha;
      // C-layout -> LDS (row M = r + 8g), to be re-read in A-frag layout
      const int row = r + (g << 3);
      P[row * 40 + ln]      = (__bf16)p0;
      P[row * 40 + 16 + ln] = (__bf16)p1;
    }
    asm volatile("" ::: "memory");  // same-wave DS ops are in-order; block compiler reordering
    const __bf16* prow = P + ln * 40;
    v8bf plo = *(const v8bf*)(prow + g * 16);
    v8bf phi = *(const v8bf*)(prow + g * 16 + 8);
    v16bf pa;
#pragma unroll
    for (int i = 0; i < 8; ++i) { pa[i] = plo[i]; pa[i + 8] = phi[i]; }
    // ---- O += P * V  (V stored transposed [D,N] so columns are contiguous)
#pragma unroll
    for (int dt = 0; dt < 4; ++dt) {
      const __bf16* vcol = Vt + (size_t)((dt << 4) + ln) * 2048 + kb;
      v16bf vb = *(const v16bf*)(vcol + (g << 4));
      O[dt] = WMMA_BF16(pa, vb, O[dt]);
    }
    asm volatile("" ::: "memory");
  }

  // epilogue: normalize, write att output in [B, N, C] bf16 (GEMM-ready rows)
  const int b_ = bh >> 4, h = bh & 15;
#pragma unroll
  for (int r = 0; r < 8; ++r) {
    const int n = qbase + r + (g << 3);
    const float inv = 1.0f / rsum[r];
    const size_t base = ((size_t)(b_ * 2048 + n)) * 1024 + h * 64 + ln;
    attb[base + 0]  = (__bf16)(O[0][r] * inv);
    attb[base + 16] = (__bf16)(O[1][r] * inv);
    attb[base + 32] = (__bf16)(O[2][r] * inv);
    attb[base + 48] = (__bf16)(O[3][r] * inv);
  }
}

// ---------------------------- output projection ----------------------------
__global__ __launch_bounds__(128) void proj_gemm(
    const __bf16* __restrict__ attb, const __bf16* __restrict__ wT,
    const float* __restrict__ bias, float* __restrict__ out) {
  const int lane = threadIdx.x & 31, wv = threadIdx.x >> 5;
  const int g = lane >> 4, ln = lane & 15;
  const int rowbase = blockIdx.x * 64;
  const int colbase = blockIdx.y * 128 + wv * 32;

  v8f acc[4][2];
#pragma unroll
  for (int ti = 0; ti < 4; ++ti)
#pragma unroll
    for (int tj = 0; tj < 2; ++tj) acc[ti][tj] = (v8f)0.0f;

  const __bf16* ar[4];
  const __bf16* bc[2];
#pragma unroll
  for (int ti = 0; ti < 4; ++ti)
    ar[ti] = attb + (size_t)(rowbase + ti * 16 + ln) * 1024;
#pragma unroll
  for (int tj = 0; tj < 2; ++tj)
    bc[tj] = wT + (size_t)(colbase + tj * 16 + ln) * 1024;

  for (int k0 = 0; k0 < 1024; k0 += 32) {
    v16bf a[4], b[2];
#pragma unroll
    for (int ti = 0; ti < 4; ++ti) a[ti] = load_a16(ar[ti], k0, g);
#pragma unroll
    for (int tj = 0; tj < 2; ++tj) b[tj] = load_b16(bc[tj], k0, g);
#pragma unroll
    for (int ti = 0; ti < 4; ++ti)
#pragma unroll
      for (int tj = 0; tj < 2; ++tj)
        acc[ti][tj] = WMMA_BF16(a[ti], b[tj], acc[ti][tj]);
  }

#pragma unroll
  for (int ti = 0; ti < 4; ++ti)
#pragma unroll
    for (int tj = 0; tj < 2; ++tj)
#pragma unroll
      for (int r = 0; r < 8; ++r) {
        const int grow = rowbase + ti * 16 + r + (g << 3);
        const int gcol = colbase + tj * 16 + ln;
        out[(size_t)grow * 1024 + gcol] = acc[ti][tj][r] + bias[gcol];
      }
}

// ------------------------------- launcher ----------------------------------
extern "C" void kernel_launch(void* const* d_in, const int* in_sizes, int n_in,
                              void* d_out, int out_size, void* d_ws, size_t ws_size,
                              hipStream_t stream) {
  (void)in_sizes; (void)n_in; (void)out_size; (void)ws_size;
  const float* x     = (const float*)d_in[0];   // [4,2048,1024]
  const float* mask  = (const float*)d_in[1];   // [2048,2048]
  const float* Wqkv  = (const float*)d_in[2];   // [1024,3072]
  const float* Wproj = (const float*)d_in[3];   // [1024,1024]
  const float* bproj = (const float*)d_in[4];   // [1024]
  float* out = (float*)d_out;

  char* ws = (char*)d_ws;
  const size_t OFF_XB    = 0;                      // 16 MB  x bf16 [8192,1024]
  const size_t OFF_WQKVT = 16777216;               //  6 MB  Wqkv^T bf16 [3072,1024]
  const size_t OFF_WPRJT = 23068672;               //  2 MB  Wproj^T bf16 [1024,1024]
  const size_t OFF_Q     = 25165824;               // 16 MB  q*1/8 bf16 [B,H,N,D]
  const size_t OFF_K     = 41943040;               // 16 MB  k bf16 [B,H,N,D]
  const size_t OFF_VT    = 58720256;               // 16 MB  v^T bf16 [B,H,D,N]
  const size_t OFF_ATT   = 75497472;               // 16 MB  att bf16 [B,N,C]

  __bf16* xb    = (__bf16*)(ws + OFF_XB);
  __bf16* wqkvT = (__bf16*)(ws + OFF_WQKVT);
  __bf16* wprjT = (__bf16*)(ws + OFF_WPRJT);
  __bf16* qd    = (__bf16*)(ws + OFF_Q);
  __bf16* kd    = (__bf16*)(ws + OFF_K);
  __bf16* vtd   = (__bf16*)(ws + OFF_VT);
  __bf16* attb  = (__bf16*)(ws + OFF_ATT);

  cvt_f32_bf16<<<2048, 256, 0, stream>>>(x, xb, 8192 * 1024);
  transpose_f32_to_bf16<<<dim3(3072 / 32, 1024 / 32), dim3(32, 8), 0, stream>>>(
      Wqkv, wqkvT, 1024, 3072);
  transpose_f32_to_bf16<<<dim3(1024 / 32, 1024 / 32), dim3(32, 8), 0, stream>>>(
      Wproj, wprjT, 1024, 1024);

  qkv_gemm<<<dim3(128, 24), 128, 0, stream>>>(xb, wqkvT, qd, kd, vtd);
  attn_kernel<<<2048, 128, 0, stream>>>(qd, kd, vtd, mask, attb);
  proj_gemm<<<dim3(128, 8), 128, 0, stream>>>(attb, wprjT, bproj, out);
}
